// PhyloEGNN_80607946211642
// MI455X (gfx1250) — compile-verified, matrix operand order checked by reading, not run on hardware
//
#include <hip/hip_runtime.h>
#include <hip/hip_bf16.h>
#include <math.h>

// ---------------------------------------------------------------------------
// EGNN on MI455X (gfx1250). Wave32, WMMA f16->f32 16x16x32.
//  - h kept as f32 [N,64] + f16 mirror h16 [N,64] (12.8MB -> L2-resident)
//  - weights repacked into FRAGMENT-MAJOR f16 layout: per (kstep,ntile)
//    fragment, each lane's 8 dwords are contiguous -> B operand = 2x b128
//  - A operand from LDS tile = 2x ds_load_b128 (CDNA5 16-bit striping is two
//    contiguous 16B runs per lane)
//  - coords double-buffered per layer to match reference .at[row].add semantics
// ---------------------------------------------------------------------------

typedef __attribute__((ext_vector_type(16))) _Float16 v16h;
typedef __attribute__((ext_vector_type(8)))  float    v8f;

#define WPB 4  // waves per block (blockDim = 128 = 4 x wave32)

union U32H2 { unsigned u; _Float16 h[2]; };
union V16HU { v16h v; unsigned u[8]; uint4 q[2]; };

__device__ __forceinline__ float silu_f(float x) { return x / (1.0f + __expf(-x)); }
__device__ __forceinline__ float sigm_f(float x) { return 1.0f / (1.0f + __expf(-x)); }

// wait for this wave's outstanding LDS ops (intra-wave staging->consume fence)
#define LDS_FENCE() asm volatile("s_wait_dscnt 0x0" ::: "memory")

__device__ __forceinline__ v8f v8f_zero() {
  v8f z = {0.f,0.f,0.f,0.f,0.f,0.f,0.f,0.f};
  return z;
}

// A fragment (16x32 f16) from an LDS tile stored row-major [16][ldk].
// CDNA5 16-bit A striping: lane&15 = row M, lane>>4 selects K half; the 8
// packed dwords are two contiguous 16B runs: halfs [kb+hf*8, +8) and +16.
__device__ __forceinline__ v16h load_a_lds(const _Float16* base, int ldk, int kb, int lane) {
  const _Float16* p = base + (lane & 15) * ldk + kb + ((lane >> 4) << 3);
  V16HU r;
  r.q[0] = *(const uint4*)p;         // ds_load_b128
  r.q[1] = *(const uint4*)(p + 16);  // ds_load_b128
  return r.v;
}

// B fragment (32x16 f16) from fragment-major packed weights:
// wf[((frag*32 + lane)*8 + v] -> two coalesced global_load_b128 per lane.
__device__ __forceinline__ v16h load_b_frag(const unsigned* __restrict__ wf,
                                            int frag, int lane) {
  const uint4* p = (const uint4*)wf + (((frag << 5) + lane) << 1);
  V16HU r;
  r.q[0] = p[0];
  r.q[1] = p[1];
  return r.v;
}

#define WMMA_F16(a, b, c) \
  __builtin_amdgcn_wmma_f32_16x16x32_f16(false, (a), false, (b), (short)0, (c), false, false)

// ------------------------- weight packing (fragment-major) -------------------------
// dst[l][ks][t][lane][v] = f16 pair (W[k0][n], W[k0+1][n]) with
// n = t*16 + (lane&15), k0 = ks*32 + (lane>>4)*8 + (v&3)*2 + (v>>2)*16.
__global__ void egnn_pack_w(const float* __restrict__ src, unsigned* __restrict__ dst,
                            int Ksrc, int ksteps, int ntiles, int Nf,
                            int lsS, int nl) {
  int per = ksteps * ntiles * 256;  // 32 lanes * 8 dwords per fragment
  int total = nl * per;
  for (int idx = blockIdx.x * blockDim.x + threadIdx.x; idx < total;
       idx += gridDim.x * blockDim.x) {
    int l = idx / per, rem = idx - l * per;
    int frag = rem >> 8;
    int r2 = rem & 255;
    int L = r2 >> 3, v = r2 & 7;
    int ks = frag / ntiles, t = frag - ks * ntiles;
    int n  = (t << 4) + (L & 15);
    int k0 = (ks << 5) + ((L >> 4) << 3) + ((v & 3) << 1) + ((v >> 2) << 4);
    float a = (k0     < Ksrc) ? src[l * lsS + k0 * Nf + n]       : 0.f;
    float b = (k0 + 1 < Ksrc) ? src[l * lsS + (k0 + 1) * Nf + n] : 0.f;
    U32H2 u; u.h[0] = (_Float16)a; u.h[1] = (_Float16)b;
    dst[(size_t)l * per + rem] = u.u;
  }
}

// ------------------------- node embedding -------------------------
__global__ void egnn_embed(const float* __restrict__ x, const float* __restrict__ in_w,
                           const float* __restrict__ in_b, float* __restrict__ h,
                           _Float16* __restrict__ h16, float* __restrict__ agg, int n) {
  size_t total = (size_t)n * 64;
  for (size_t idx = blockIdx.x * (size_t)blockDim.x + threadIdx.x; idx < total;
       idx += (size_t)gridDim.x * blockDim.x) {
    int k = (int)(idx & 63);
    size_t i = idx >> 6;
    float v = x[i] * in_w[k] + in_b[k];
    h[idx] = v;
    h16[idx] = (_Float16)v;
    agg[idx] = 0.f;
  }
}

// ------------------------- coord phase -------------------------
struct __align__(16) CoordShm {
  _Float16 astage[16 * 160];  // 16 edges x (h_row64 | h_col64 | rel3 pad->160)
  _Float16 hid[16 * 136];     // 16 x 128 hidden, padded stride
  int srow[16];
  int scol[16];
  float sdelta[48];
};

__global__ __launch_bounds__(128) void egnn_coord(
    const _Float16* __restrict__ h16, const float* __restrict__ ccur,
    float* __restrict__ cnext, const int* __restrict__ erow,
    const int* __restrict__ ecol, const unsigned* __restrict__ w1f,
    const float* __restrict__ b1, const float* __restrict__ w2,
    const float* __restrict__ b2, const float* __restrict__ eww,
    const float* __restrict__ ewbp, const float* __restrict__ cnsp, int nEdges) {
  __shared__ CoordShm shm[WPB];
  int wave = threadIdx.x >> 5, lane = threadIdx.x & 31;
  CoordShm& S = shm[wave];
  int ccolv = lane & 15, hf = lane >> 4;
  int ntiles = nEdges >> 4;
  for (int tile = blockIdx.x * WPB + wave; tile < ntiles; tile += gridDim.x * WPB) {
    int ebase = tile << 4;
    if (lane < 16) { S.srow[lane] = erow[ebase + lane]; S.scol[lane] = ecol[ebase + lane]; }
    LDS_FENCE();
    for (int i = lane; i < 256; i += 32) {        // gather h rows, 16B chunks
      int e = i >> 4, p = i & 15;
      int node = (p < 8) ? S.srow[e] : S.scol[e];
      *(uint4*)(&S.astage[e * 160 + p * 8]) =
          ((const uint4*)(h16 + (size_t)node * 64))[p & 7];
    }
    if (lane < 16) {                               // rel (OLD coords) + zero pad
      int r = S.srow[lane], c = S.scol[lane];
      _Float16* d = &S.astage[lane * 160 + 128];
      d[0] = (_Float16)(ccur[r * 3 + 0] - ccur[c * 3 + 0]);
      d[1] = (_Float16)(ccur[r * 3 + 1] - ccur[c * 3 + 1]);
      d[2] = (_Float16)(ccur[r * 3 + 2] - ccur[c * 3 + 2]);
#pragma unroll
      for (int k = 3; k < 32; ++k) d[k] = (_Float16)0.f;
    }
    LDS_FENCE();
    // GEMM1: [16x160]x[160x128] -> silu -> hid
    v8f acc[8];
#pragma unroll
    for (int t = 0; t < 8; ++t) acc[t] = v8f_zero();
#pragma unroll
    for (int kk = 0; kk < 5; ++kk) {
      v16h a = load_a_lds(S.astage, 160, kk * 32, lane);
#pragma unroll
      for (int t = 0; t < 8; ++t) {
        v16h b = load_b_frag(w1f, kk * 8 + t, lane);
        acc[t] = WMMA_F16(a, b, acc[t]);
      }
    }
#pragma unroll
    for (int t = 0; t < 8; ++t) {
      float bias = b1[t * 16 + ccolv];
#pragma unroll
      for (int r = 0; r < 8; ++r)
        S.hid[(hf * 8 + r) * 136 + t * 16 + ccolv] = (_Float16)silu_f(acc[t][r] + bias);
    }
    LDS_FENCE();
    // tiny head [128,3]: scalar dots (WMMA N=3 would waste 13/16 of the tile)
    for (int j = lane; j < 48; j += 32) {
      int e = j / 3, c = j - e * 3;
      float s = b2[c];
#pragma unroll 8
      for (int k = 0; k < 128; ++k) s += (float)S.hid[e * 136 + k] * w2[k * 3 + c];
      S.sdelta[j] = s;
    }
    LDS_FENCE();
    if (lane < 16) {  // CoorsNorm + sigmoid gate + scatter into NEXT coords
      float dx = S.sdelta[lane * 3 + 0], dy = S.sdelta[lane * 3 + 1], dz = S.sdelta[lane * 3 + 2];
      float nrm = fmaxf(sqrtf(dx * dx + dy * dy + dz * dz), 1e-8f);
      int r = S.srow[lane], c = S.scol[lane];
      float rx = ccur[r * 3 + 0] - ccur[c * 3 + 0];
      float ry = ccur[r * 3 + 1] - ccur[c * 3 + 1];
      float rz = ccur[r * 3 + 2] - ccur[c * 3 + 2];
      float gate = sigm_f(rx * eww[0] + ry * eww[1] + rz * eww[2] + ewbp[0]);
      float f = (cnsp[0] / nrm) * gate;
      atomicAdd(&cnext[r * 3 + 0], dx * f);
      atomicAdd(&cnext[r * 3 + 1], dy * f);
      atomicAdd(&cnext[r * 3 + 2], dz * f);
    }
    LDS_FENCE();
  }
}

// ------------------------- edge message phase -------------------------
struct __align__(16) EdgeShm {
  _Float16 astage[16 * 160];
  _Float16 hid[16 * 136];
  int srow[16];
  int scol[16];
};

__global__ __launch_bounds__(128) void egnn_edge(
    const _Float16* __restrict__ h16, const float* __restrict__ coords,
    const int* __restrict__ erow, const int* __restrict__ ecol,
    const unsigned* __restrict__ w1f, const float* __restrict__ b1,
    const unsigned* __restrict__ w2f, const float* __restrict__ b2,
    float* __restrict__ agg, int nEdges) {
  __shared__ EdgeShm shm[WPB];
  int wave = threadIdx.x >> 5, lane = threadIdx.x & 31;
  EdgeShm& S = shm[wave];
  int ccolv = lane & 15, hf = lane >> 4;
  int ntiles = nEdges >> 4;
  for (int tile = blockIdx.x * WPB + wave; tile < ntiles; tile += gridDim.x * WPB) {
    int ebase = tile << 4;
    if (lane < 16) { S.srow[lane] = erow[ebase + lane]; S.scol[lane] = ecol[ebase + lane]; }
    LDS_FENCE();
    for (int i = lane; i < 256; i += 32) {
      int e = i >> 4, p = i & 15;
      int node = (p < 8) ? S.srow[e] : S.scol[e];
      *(uint4*)(&S.astage[e * 160 + p * 8]) =
          ((const uint4*)(h16 + (size_t)node * 64))[p & 7];
    }
    if (lane < 16) {                               // rel from NEW coords
      int r = S.srow[lane], c = S.scol[lane];
      _Float16* d = &S.astage[lane * 160 + 128];
      d[0] = (_Float16)(coords[r * 3 + 0] - coords[c * 3 + 0]);
      d[1] = (_Float16)(coords[r * 3 + 1] - coords[c * 3 + 1]);
      d[2] = (_Float16)(coords[r * 3 + 2] - coords[c * 3 + 2]);
#pragma unroll
      for (int k = 3; k < 32; ++k) d[k] = (_Float16)0.f;
    }
    LDS_FENCE();
    v8f acc[8];
#pragma unroll
    for (int t = 0; t < 8; ++t) acc[t] = v8f_zero();
#pragma unroll
    for (int kk = 0; kk < 5; ++kk) {
      v16h a = load_a_lds(S.astage, 160, kk * 32, lane);
#pragma unroll
      for (int t = 0; t < 8; ++t) {
        v16h b = load_b_frag(w1f, kk * 8 + t, lane);
        acc[t] = WMMA_F16(a, b, acc[t]);
      }
    }
#pragma unroll
    for (int t = 0; t < 8; ++t) {
      float bias = b1[t * 16 + ccolv];
#pragma unroll
      for (int r = 0; r < 8; ++r)
        S.hid[(hf * 8 + r) * 136 + t * 16 + ccolv] = (_Float16)silu_f(acc[t][r] + bias);
    }
    LDS_FENCE();
    // GEMM2: [16x128]x[128x64] -> silu -> scatter-add into agg[row]
    v8f acc2[4];
#pragma unroll
    for (int t = 0; t < 4; ++t) acc2[t] = v8f_zero();
#pragma unroll
    for (int kk = 0; kk < 4; ++kk) {
      v16h a = load_a_lds(S.hid, 136, kk * 32, lane);
#pragma unroll
      for (int t = 0; t < 4; ++t) {
        v16h b = load_b_frag(w2f, kk * 4 + t, lane);
        acc2[t] = WMMA_F16(a, b, acc2[t]);
      }
    }
    int myrow[8];
#pragma unroll
    for (int r = 0; r < 8; ++r) myrow[r] = S.srow[hf * 8 + r];
#pragma unroll
    for (int t = 0; t < 4; ++t) {
      float bias = b2[t * 16 + ccolv];
#pragma unroll
      for (int r = 0; r < 8; ++r) {
        float v = silu_f(acc2[t][r] + bias);
        atomicAdd(&agg[(size_t)myrow[r] * 64 + t * 16 + ccolv], v);  // L2-resident
      }
    }
    LDS_FENCE();
  }
}

// ------------------------- node update phase -------------------------
struct __align__(16) NodeShm {
  _Float16 astage[16 * 128];  // [h || agg]
  _Float16 hid[16 * 136];
  float hn[16 * 68];          // h + upd (pre-layernorm)
};

__global__ __launch_bounds__(128) void egnn_node(
    float* __restrict__ h, _Float16* __restrict__ h16, float* __restrict__ agg,
    const unsigned* __restrict__ w1f, const float* __restrict__ b1,
    const unsigned* __restrict__ w2f, const float* __restrict__ b2, int nNodes) {
  __shared__ NodeShm shm[WPB];
  int wave = threadIdx.x >> 5, lane = threadIdx.x & 31;
  NodeShm& S = shm[wave];
  int ccolv = lane & 15, hf = lane >> 4;
  int ntiles = nNodes >> 4;
  for (int tile = blockIdx.x * WPB + wave; tile < ntiles; tile += gridDim.x * WPB) {
    int base = tile << 4;
    for (int i = lane; i < 128; i += 32) {         // stage h16
      int e = i >> 3, p = i & 7;
      *(uint4*)(&S.astage[e * 128 + p * 8]) =
          ((const uint4*)(h16 + (size_t)(base + e) * 64))[p];
    }
    for (int i = lane; i < 1024; i += 32) {        // stage agg (f32->f16)
      int e = i >> 6, k = i & 63;
      S.astage[e * 128 + 64 + k] = (_Float16)agg[(size_t)(base + e) * 64 + k];
    }
    LDS_FENCE();
    v8f acc[8];
#pragma unroll
    for (int t = 0; t < 8; ++t) acc[t] = v8f_zero();
#pragma unroll
    for (int kk = 0; kk < 4; ++kk) {
      v16h a = load_a_lds(S.astage, 128, kk * 32, lane);
#pragma unroll
      for (int t = 0; t < 8; ++t) {
        v16h b = load_b_frag(w1f, kk * 8 + t, lane);
        acc[t] = WMMA_F16(a, b, acc[t]);
      }
    }
#pragma unroll
    for (int t = 0; t < 8; ++t) {
      float bias = b1[t * 16 + ccolv];
#pragma unroll
      for (int r = 0; r < 8; ++r)
        S.hid[(hf * 8 + r) * 136 + t * 16 + ccolv] = (_Float16)silu_f(acc[t][r] + bias);
    }
    LDS_FENCE();
    v8f acc2[4];
#pragma unroll
    for (int t = 0; t < 4; ++t) acc2[t] = v8f_zero();
#pragma unroll
    for (int kk = 0; kk < 4; ++kk) {
      v16h a = load_a_lds(S.hid, 136, kk * 32, lane);
#pragma unroll
      for (int t = 0; t < 4; ++t) {
        v16h b = load_b_frag(w2f, kk * 4 + t, lane);
        acc2[t] = WMMA_F16(a, b, acc2[t]);
      }
    }
#pragma unroll
    for (int t = 0; t < 4; ++t) {
      float bias = b2[t * 16 + ccolv];   // no silu on second GEMM (reference)
#pragma unroll
      for (int r = 0; r < 8; ++r) {
        int m = hf * 8 + r, col = t * 16 + ccolv;
        S.hn[m * 68 + col] = h[(size_t)(base + m) * 64 + col] + acc2[t][r] + bias;
      }
    }
    LDS_FENCE();
    if (lane < 16) {  // layernorm row, writeback, zero agg for next layer
      size_t node = (size_t)(base + lane);
      float mu = 0.f;
#pragma unroll 8
      for (int k = 0; k < 64; ++k) mu += S.hn[lane * 68 + k];
      mu *= (1.f / 64.f);
      float var = 0.f;
#pragma unroll 8
      for (int k = 0; k < 64; ++k) { float d = S.hn[lane * 68 + k] - mu; var += d * d; }
      var *= (1.f / 64.f);
      float inv = rsqrtf(var + 1e-5f);
#pragma unroll 8
      for (int k = 0; k < 64; ++k) {
        float o = (S.hn[lane * 68 + k] - mu) * inv;
        h[node * 64 + k] = o;
        h16[node * 64 + k] = (_Float16)o;
        agg[node * 64 + k] = 0.f;
      }
    }
    LDS_FENCE();
  }
}

// ------------------------- pooling + output head -------------------------
__global__ void egnn_pool_init(float* sums, float* cnts) {
  int t = blockIdx.x * blockDim.x + threadIdx.x;
  if (t < 64 * 64) sums[t] = 0.f;
  if (t < 64) cnts[t] = 0.f;
}

__global__ void egnn_pool_sum(const float* __restrict__ h, const int* __restrict__ batch,
                              float* __restrict__ sums, float* __restrict__ cnts, int n) {
  size_t total = (size_t)n * 64;
  for (size_t idx = blockIdx.x * (size_t)blockDim.x + threadIdx.x; idx < total;
       idx += (size_t)gridDim.x * blockDim.x) {
    size_t i = idx >> 6;
    int k = (int)(idx & 63);
    int g = batch[i];
    atomicAdd(&sums[g * 64 + k], h[idx]);
    if (k == 0) atomicAdd(&cnts[g], 1.f);
  }
}

__global__ __launch_bounds__(256) void egnn_out(const float* __restrict__ sums,
                                                const float* __restrict__ cnts,
                                                const float* __restrict__ out_w,
                                                const float* __restrict__ out_b,
                                                float* __restrict__ out) {
  __shared__ float so[64 * 128];
  __shared__ float snorm[64];
  int tid = threadIdx.x;
  for (int idx = tid; idx < 8192; idx += 256) {
    int g = idx >> 7, j = idx & 127;
    float c = fmaxf(cnts[g], 1.f);
    float s = out_b[j];
#pragma unroll 8
    for (int k = 0; k < 64; ++k) {
      float p = fmaxf(sums[g * 64 + k] / c, 0.f);
      s += p * out_w[k * 128 + j];
    }
    so[idx] = s;
  }
  __syncthreads();
  if (tid < 64) {
    float s = 0.f;
    for (int j = 0; j < 128; ++j) { float v = so[tid * 128 + j]; s += v * v; }
    snorm[tid] = fmaxf(sqrtf(s), 1e-12f);
  }
  __syncthreads();
  for (int idx = tid; idx < 8192; idx += 256) out[idx] = so[idx] / snorm[idx >> 7];
}

// ------------------------- host orchestration -------------------------
extern "C" void kernel_launch(void* const* d_in, const int* in_sizes, int n_in,
                              void* d_out, int out_size, void* d_ws, size_t ws_size,
                              hipStream_t stream) {
  const int N = 100000, E = 1000000;
  const float* x        = (const float*)d_in[0];
  const float* coords_i = (const float*)d_in[1];
  const float* in_w     = (const float*)d_in[2];
  const float* in_b     = (const float*)d_in[3];
  const float* coord_w1 = (const float*)d_in[4];
  const float* coord_b1 = (const float*)d_in[5];
  const float* coord_w2 = (const float*)d_in[6];
  const float* coord_b2 = (const float*)d_in[7];
  const float* ew_w     = (const float*)d_in[8];
  const float* ew_b     = (const float*)d_in[9];
  const float* cn_scale = (const float*)d_in[10];
  const float* edge_w1  = (const float*)d_in[11];
  const float* edge_b1  = (const float*)d_in[12];
  const float* edge_w2  = (const float*)d_in[13];
  const float* edge_b2  = (const float*)d_in[14];
  const float* node_w1  = (const float*)d_in[15];
  const float* node_b1  = (const float*)d_in[16];
  const float* node_w2  = (const float*)d_in[17];
  const float* node_b2  = (const float*)d_in[18];
  const float* out_w    = (const float*)d_in[19];
  const float* out_b    = (const float*)d_in[20];
  const int*   eidx     = (const int*)d_in[21];
  const int*   batch    = (const int*)d_in[22];
  (void)in_sizes; (void)n_in; (void)out_size; (void)ws_size;

  char* ws = (char*)d_ws;
  size_t off = 0;
  auto take = [&](size_t bytes) -> char* {
    char* p = ws + off;
    off = (off + bytes + 255) & ~(size_t)255;
    return p;
  };
  float*     coordsA = (float*)take((size_t)N * 3 * 4);
  float*     coordsB = (float*)take((size_t)N * 3 * 4);
  float*     h       = (float*)take((size_t)N * 64 * 4);
  _Float16*  h16     = (_Float16*)take((size_t)N * 64 * 2);
  float*     agg     = (float*)take((size_t)N * 64 * 4);
  float*     sums    = (float*)take(64 * 64 * 4);
  float*     cnts    = (float*)take(64 * 4);
  // fragment-major packed weights (dword counts per layer):
  const int C1 = 5 * 8 * 256;  // Kpad160 x N128 : 10240
  const int E2 = 4 * 4 * 256;  // Kpad128 x N64  : 4096
  const int N1 = 4 * 8 * 256;  // Kpad128 x N128 : 8192
  unsigned* cw1f = (unsigned*)take((size_t)4 * C1 * 4);
  unsigned* ew1f = (unsigned*)take((size_t)4 * C1 * 4);
  unsigned* ew2f = (unsigned*)take((size_t)4 * E2 * 4);
  unsigned* nw1f = (unsigned*)take((size_t)4 * N1 * 4);
  unsigned* nw2f = (unsigned*)take((size_t)4 * E2 * 4);

  // pack weights into WMMA fragment-major layout (f16, K zero-padded)
  egnn_pack_w<<<160, 256, 0, stream>>>(coord_w1, cw1f, 131, 5, 8, 128, 131 * 128, 4);
  egnn_pack_w<<<160, 256, 0, stream>>>(edge_w1,  ew1f, 131, 5, 8, 128, 131 * 128, 4);
  egnn_pack_w<<< 64, 256, 0, stream>>>(edge_w2,  ew2f, 128, 4, 4,  64, 128 * 64,  4);
  egnn_pack_w<<<128, 256, 0, stream>>>(node_w1,  nw1f, 128, 4, 8, 128, 128 * 128, 4);
  egnn_pack_w<<< 64, 256, 0, stream>>>(node_w2,  nw2f, 128, 4, 4,  64, 128 * 64,  4);

  // h = x @ in_w + in_b ; agg = 0 ; h16 mirror
  egnn_embed<<<12800, 256, 0, stream>>>(x, in_w, in_b, h, h16, agg, N);

  // coords working copy (inputs must not be mutated)
  hipMemcpyAsync(coordsA, coords_i, (size_t)N * 3 * 4, hipMemcpyDeviceToDevice, stream);

  const int eTiles = E / 16;                 // 62500
  const int eBlocks = eTiles / WPB;          // 15625
  const int nTiles = N / 16;                 // 6250
  const int nBlocks = (nTiles + WPB - 1) / WPB;

  float* cur = coordsA;
  float* nxt = coordsB;
  for (int l = 0; l < 4; ++l) {
    // double-buffer coords: all deltas from OLD coords, scatter into NEW copy
    hipMemcpyAsync(nxt, cur, (size_t)N * 3 * 4, hipMemcpyDeviceToDevice, stream);
    egnn_coord<<<eBlocks, 128, 0, stream>>>(
        h16, cur, nxt, eidx, eidx + E,
        cw1f + (size_t)l * C1, coord_b1 + l * 128,
        coord_w2 + (size_t)l * 128 * 3, coord_b2 + l * 3,
        ew_w + l * 3, ew_b + l, cn_scale + l, E);
    egnn_edge<<<eBlocks, 128, 0, stream>>>(
        h16, nxt, eidx, eidx + E,
        ew1f + (size_t)l * C1, edge_b1 + l * 128,
        ew2f + (size_t)l * E2, edge_b2 + l * 64, agg, E);
    egnn_node<<<nBlocks, 128, 0, stream>>>(
        h, h16, agg,
        nw1f + (size_t)l * N1, node_b1 + l * 128,
        nw2f + (size_t)l * E2, node_b2 + l * 64, N);
    float* t = cur; cur = nxt; nxt = t;
  }

  egnn_pool_init<<<(64 * 64 + 255) / 256, 256, 0, stream>>>(sums, cnts);
  egnn_pool_sum<<<8192, 256, 0, stream>>>(h, batch, sums, cnts, N);
  egnn_out<<<1, 256, 0, stream>>>(sums, cnts, out_w, out_b, (float*)d_out);
}
